// RefineHead_27324581937578
// MI455X (gfx1250) — compile-verified, matrix operand order checked by reading, not run on hardware
//
#include <hip/hip_runtime.h>
#include <hip/hip_bf16.h>

// ---------------- problem constants ----------------
constexpr int kB  = 16;
constexpr int kN  = 512;
constexpr int kS  = 36;
constexpr int kC  = 64;
constexpr int kFC = 192;
constexpr int kG  = 6;
constexpr int kNOFF = 72;
constexpr int kPRI  = 4 + kNOFF;     // 76 floats per prior row
constexpr int kTOK  = kB * kN;       // 8192 tokens
constexpr int kKIN  = kS * kC;       // 2304 sampled features per token
constexpr int kD    = kFC / kG;      // 32 head dim

typedef __attribute__((ext_vector_type(16))) _Float16 v16h;
typedef __attribute__((ext_vector_type(8)))  _Float16 v8h;
typedef __attribute__((ext_vector_type(8)))  float    v8f;

// ---------------- fp32 -> fp16 convert ----------------
__global__ void cvt_f16_kernel(const float* __restrict__ src, _Float16* __restrict__ dst, int n) {
    int i = blockIdx.x * blockDim.x + threadIdx.x;
    if (i < n) dst[i] = (_Float16)src[i];
}

// ---------------- grid sample + level softmax ----------------
__device__ inline float bilin_tap(const float* base, int W, int H, float xi, float yi) {
    bool valid = (xi >= 0.f) && (xi <= (float)(W - 1)) && (yi >= 0.f) && (yi <= (float)(H - 1));
    int xc = (int)fminf(fmaxf(xi, 0.f), (float)(W - 1));
    int yc = (int)fminf(fmaxf(yi, 0.f), (float)(H - 1));
    float v = base[yc * W + xc];
    return valid ? v : 0.f;
}

__device__ inline float bilinear(const float* f, int b, int c, int H, int W, float gx, float gy) {
    float xf = (gx + 1.f) * 0.5f * (float)(W - 1);
    float yf = (gy + 1.f) * 0.5f * (float)(H - 1);
    float x0 = floorf(xf), y0 = floorf(yf);
    float wx = xf - x0, wy = yf - y0;
    const float* base = f + ((size_t)b * kC + c) * (size_t)(H * W);
    float v00 = bilin_tap(base, W, H, x0,       y0);
    float v10 = bilin_tap(base, W, H, x0 + 1.f, y0);
    float v01 = bilin_tap(base, W, H, x0,       y0 + 1.f);
    float v11 = bilin_tap(base, W, H, x0 + 1.f, y0 + 1.f);
    return v00 * (1.f - wx) * (1.f - wy) + v10 * wx * (1.f - wy)
         + v01 * (1.f - wx) * wy         + v11 * wx * wy;
}

__global__ void sample_kernel(const float* __restrict__ f0, const float* __restrict__ f1,
                              const float* __restrict__ f2, const float* __restrict__ priors,
                              const float* __restrict__ z_emb, _Float16* __restrict__ A_feat) {
    int bn = blockIdx.x;              // token 0..8191
    int b  = bn >> 9;                 // / 512
    const float* prior = priors + (size_t)bn * kPRI;
    for (int idx = threadIdx.x; idx < kS * kC; idx += blockDim.x) {
        int s = idx >> 6;
        int c = idx & 63;
        int j  = (kS - 1) - s;
        int sx = (j * 71) / 35;       // floor(linspace(0,1,36)[j] * 71)
        float gx = prior[4 + sx] * 2.f - 1.f;
        float py = 1.f - (float)sx * (1.f / 71.f);
        float gy = py * 2.f - 1.f;
        float z  = z_emb[s];
        float e0 = __expf(-0.5f * z * z);
        float e1 = __expf(-0.5f * (z - 1.f) * (z - 1.f));
        float e2 = __expf(-0.5f * (z - 2.f) * (z - 2.f));
        float inv = 1.f / (e0 + e1 + e2);
        float acc = e0 * inv * bilinear(f0, b, c, 40, 100, gx, gy)
                  + e1 * inv * bilinear(f1, b, c, 20,  50, gx, gy)
                  + e2 * inv * bilinear(f2, b, c, 10,  25, gx, gy);
        A_feat[(size_t)bn * kKIN + idx] = (_Float16)acc;
    }
}

// ---------------- WMMA fragment helpers ----------------
// A/B fragment: 16x32 f16 tile, row-major source with leading dim ld.
// Per CDNA5 ISA: lanes 0-15 hold rows M=0..15 with K = {0..7, 16..23};
// lanes 16-31 hold the same rows with K = {8..15, 24..31}.
__device__ inline v16h load_frag16x32(const _Float16* __restrict__ base, int ld, int rowMax) {
    int lane = threadIdx.x & 31;
    int r  = lane & 15;
    if (r > rowMax) r = rowMax;       // clamp for ragged N tiles (results masked at store)
    int hv = lane >> 4;
    const _Float16* p = base + (size_t)r * ld + hv * 8;
    v8h lo = *(const v8h*)(p);        // K = hv*8 .. hv*8+7
    v8h hi = *(const v8h*)(p + 16);   // K = 16+hv*8 .. 16+hv*8+7
    v16h out;
#pragma unroll
    for (int i = 0; i < 8; ++i) { out[i] = lo[i]; out[i + 8] = hi[i]; }
    return out;
}

// 32x32 output tile per wave: c[mi][ni] covers rows m0+16*mi, cols n0+16*ni.
// out[m][n] = sum_k A[m][k] * B[n][k]  (B row-per-output-col, i.e. torch (out,in)).
// hasN1 / nMax are wave-uniform, so EXEC stays all-ones at every WMMA.
__device__ inline void wmma_core2x2(const _Float16* __restrict__ A, int lda,
                                    const _Float16* __restrict__ B0,
                                    const _Float16* __restrict__ B1, int ldb,
                                    int K, int nMax0, int nMax1, bool hasN1,
                                    v8f c[2][2]) {
    for (int k = 0; k < K; k += 32) {
        if (k + 32 < K) {
            __builtin_prefetch(A + k + 32, 0, 1);   // gfx1250 global_prefetch
        }
        v16h a0 = load_frag16x32(A + k, lda, 15);
        v16h a1 = load_frag16x32(A + (size_t)16 * lda + k, lda, 15);
        v16h b0 = load_frag16x32(B0 + k, ldb, nMax0);
        c[0][0] = __builtin_amdgcn_wmma_f32_16x16x32_f16(false, a0, false, b0,
                                                         (short)0, c[0][0], false, false);
        c[1][0] = __builtin_amdgcn_wmma_f32_16x16x32_f16(false, a1, false, b0,
                                                         (short)0, c[1][0], false, false);
        if (hasN1) {
            v16h b1 = load_frag16x32(B1 + k, ldb, nMax1);
            c[0][1] = __builtin_amdgcn_wmma_f32_16x16x32_f16(false, a0, false, b1,
                                                             (short)0, c[0][1], false, false);
            c[1][1] = __builtin_amdgcn_wmma_f32_16x16x32_f16(false, a1, false, b1,
                                                             (short)0, c[1][1], false, false);
        }
    }
}

// ---------------- grouped gather GEMM (32x32 tiles; one group per n-tile) ----------------
__global__ void gather_gemm(const _Float16* __restrict__ A_feat, const _Float16* __restrict__ gw16,
                            const float* __restrict__ gather_b,
                            float* __restrict__ feat32, _Float16* __restrict__ feat16) {
    int m0 = blockIdx.x * 32;
    int n0 = blockIdx.y * 32;
    int g  = n0 >> 5;                 // group = output col / 32 (tile == one group)
    v8f c[2][2] = {};
    wmma_core2x2(A_feat + (size_t)m0 * kKIN + g * 384, kKIN,
                 gw16 + (size_t)n0 * 384, gw16 + (size_t)(n0 + 16) * 384, 384,
                 384, 15, 15, true, c);
    int lane = threadIdx.x & 31, hv = lane >> 4, nl = lane & 15;
#pragma unroll
    for (int ni = 0; ni < 2; ++ni) {
        int gn = n0 + ni * 16 + nl;
        float bv = gather_b[gn];
#pragma unroll
        for (int mi = 0; mi < 2; ++mi) {
#pragma unroll
            for (int j = 0; j < 8; ++j) {
                int gm = m0 + mi * 16 + hv * 8 + j;
                float v = c[mi][ni][j] + bv;
                feat32[(size_t)gm * kFC + gn] = v;
                feat16[(size_t)gm * kFC + gn] = (_Float16)v;
            }
        }
    }
}

// ---------------- generic linear (WMMA) with optional relu/residual/f16/transposed-V store ----
__global__ void lin_gemm(const _Float16* __restrict__ A, int lda,
                         const _Float16* __restrict__ W, int ldb,
                         const float* __restrict__ bias,
                         int Nn, int K, int relu, int vt,
                         float* __restrict__ out32, int ldo,
                         _Float16* __restrict__ out16, int ldo16,
                         const float* __restrict__ resid) {
    int m0 = blockIdx.x * 32;
    int n0 = blockIdx.y * 32;
    bool hasN1 = (n0 + 16) < Nn;
    int nMax0 = Nn - 1 - n0;        if (nMax0 > 15) nMax0 = 15;
    int nMax1 = Nn - 1 - (n0 + 16); if (nMax1 > 15) nMax1 = 15;
    if (nMax1 < 0) nMax1 = 0;
    v8f c[2][2] = {};
    wmma_core2x2(A + (size_t)m0 * lda, lda,
                 W + (size_t)n0 * ldb, W + (size_t)(n0 + 16) * ldb, ldb,
                 K, nMax0, nMax1, hasN1, c);
    int lane = threadIdx.x & 31, hv = lane >> 4, nl = lane & 15;
#pragma unroll
    for (int ni = 0; ni < 2; ++ni) {
        int gn = n0 + ni * 16 + nl;
        bool nOK = gn < Nn;
        float bv = (bias != nullptr && nOK) ? bias[gn] : 0.f;
#pragma unroll
        for (int mi = 0; mi < 2; ++mi) {
#pragma unroll
            for (int j = 0; j < 8; ++j) {
                int gm = m0 + mi * 16 + hv * 8 + j;
                float v = c[mi][ni][j] + bv;
                if (relu) v = fmaxf(v, 0.f);
                if (nOK) {
                    if (resid) v += resid[(size_t)gm * ldo + gn];
                    if (out32) out32[(size_t)gm * ldo + gn] = v;
                    if (out16) {
                        if (vt)   // store V transposed: [b][n][token-in-b], token rows of 512
                            out16[((size_t)((gm >> 9) * kFC + gn) << 9) + (gm & 511)] = (_Float16)v;
                        else
                            out16[(size_t)gm * ldo16 + gn] = (_Float16)v;
                    }
                }
            }
        }
    }
}

// ---------------- attention scores: per (b,g) 512x512x32 ----------------
__global__ void scores_gemm(const _Float16* __restrict__ q16, const _Float16* __restrict__ k16,
                            float* __restrict__ scores) {
    int bg = blockIdx.z;
    int b = bg / kG, g = bg % kG;
    int m0 = blockIdx.x * 32;   // query index n
    int n0 = blockIdx.y * 32;   // key index m
    v8f c[2][2] = {};
    const _Float16* kbase = k16 + (size_t)(b * kN) * kFC + g * kD;
    wmma_core2x2(q16 + (size_t)(b * kN + m0) * kD, kD,
                 kbase + (size_t)n0 * kFC, kbase + (size_t)(n0 + 16) * kFC, kFC,
                 kD, 15, 15, true, c);
    int lane = threadIdx.x & 31, hv = lane >> 4, nl = lane & 15;
    float* out = scores + (size_t)bg * kN * kN;
    const float scale = 0.17677669529663688f;    // 32^-0.5
#pragma unroll
    for (int mi = 0; mi < 2; ++mi)
#pragma unroll
        for (int ni = 0; ni < 2; ++ni)
#pragma unroll
            for (int j = 0; j < 8; ++j) {
                int gm = m0 + mi * 16 + hv * 8 + j;
                int gn = n0 + ni * 16 + nl;
                out[(size_t)gm * kN + gn] = c[mi][ni][j] * scale;
            }
}

// ---------------- softmax over last dim (512), in-place fp32 + f16 copy ----------------
__global__ void softmax_kernel(float* __restrict__ scores, _Float16* __restrict__ attn16) {
    __shared__ float red[256];
    size_t row = blockIdx.x;
    float* p = scores + row * kN;
    _Float16* p16 = attn16 + row * kN;
    int t = threadIdx.x;
    float a = p[t];
    float bb = p[t + 256];
    red[t] = fmaxf(a, bb);
    __syncthreads();
    for (int o = 128; o > 0; o >>= 1) {
        if (t < o) red[t] = fmaxf(red[t], red[t + o]);
        __syncthreads();
    }
    float mx = red[0];
    __syncthreads();
    float ea = __expf(a - mx);
    float eb = __expf(bb - mx);
    red[t] = ea + eb;
    __syncthreads();
    for (int o = 128; o > 0; o >>= 1) {
        if (t < o) red[t] += red[t + o];
        __syncthreads();
    }
    float inv = 1.f / red[0];
    p[t]        = ea * inv;
    p[t + 256]  = eb * inv;
    p16[t]       = (_Float16)(ea * inv);
    p16[t + 256] = (_Float16)(eb * inv);
}

// ---------------- ctx: per (b,g) 512x32x512 ----------------
__global__ void ctx_gemm(const _Float16* __restrict__ attn16, const _Float16* __restrict__ vT16,
                         _Float16* __restrict__ ctx16) {
    int bg = blockIdx.z;
    int b = bg / kG, g = bg % kG;
    int m0 = blockIdx.x * 32;   // token n within batch b
    v8f c[2][2] = {};
    const _Float16* vbase = vT16 + (size_t)(b * kFC + g * kD) * kN;
    wmma_core2x2(attn16 + (size_t)bg * kN * kN + (size_t)m0 * kN, kN,
                 vbase, vbase + (size_t)16 * kN, kN,
                 kN, 15, 15, true, c);
    int lane = threadIdx.x & 31, hv = lane >> 4, nl = lane & 15;
#pragma unroll
    for (int mi = 0; mi < 2; ++mi)
#pragma unroll
        for (int ni = 0; ni < 2; ++ni)
#pragma unroll
            for (int j = 0; j < 8; ++j) {
                int gm = m0 + mi * 16 + hv * 8 + j;
                ctx16[(size_t)(b * kN + gm) * kFC + g * kD + ni * 16 + nl] = (_Float16)c[mi][ni][j];
            }
}

// ---------------- final assembly ----------------
__global__ void assemble_kernel(const float* __restrict__ cls32, const float* __restrict__ reg32,
                                const float* __restrict__ priors, float* __restrict__ pred) {
    int i = blockIdx.x * blockDim.x + threadIdx.x;
    if (i >= kTOK * kPRI) return;
    int bn = i / kPRI, o = i % kPRI;
    float v;
    if (o < 2) v = cls32[bn * 2 + o];
    else       v = priors[(size_t)bn * kPRI + o] + reg32[(size_t)bn * 74 + (o - 2)];
    pred[i] = v;
}

// ---------------- host launch ----------------
extern "C" void kernel_launch(void* const* d_in, const int* in_sizes, int n_in,
                              void* d_out, int out_size, void* d_ws, size_t ws_size,
                              hipStream_t stream) {
    const float* feat0    = (const float*)d_in[0];
    const float* feat1    = (const float*)d_in[1];
    const float* feat2    = (const float*)d_in[2];
    const float* priors   = (const float*)d_in[3];
    const float* z_emb    = (const float*)d_in[4];
    const float* gather_w = (const float*)d_in[5];
    const float* gather_b = (const float*)d_in[6];
    const float* q_w      = (const float*)d_in[7];
    const float* k_w      = (const float*)d_in[8];
    const float* v_w      = (const float*)d_in[9];
    const float* ch1_w    = (const float*)d_in[10];
    const float* ch1_b    = (const float*)d_in[11];
    const float* ch2_w    = (const float*)d_in[12];
    const float* ch2_b    = (const float*)d_in[13];
    const float* cls_m_w  = (const float*)d_in[14];
    const float* cls_m_b  = (const float*)d_in[15];
    const float* cls_w    = (const float*)d_in[16];
    const float* cls_b    = (const float*)d_in[17];
    const float* reg_m_w  = (const float*)d_in[18];
    const float* reg_m_b  = (const float*)d_in[19];
    const float* reg_w    = (const float*)d_in[20];
    const float* reg_b    = (const float*)d_in[21];

    float* out      = (float*)d_out;
    float* pred_out = out;                                    // 16*512*76
    float* fc_out   = out + (size_t)kTOK * kPRI;              // 8192*192
    float* attn_out = fc_out + (size_t)kTOK * kFC;            // 96*512*512

    char* ws = (char*)d_ws;
    size_t off = 0;
    auto alloc = [&](size_t bytes) -> void* {
        void* p = ws + off;
        off += bytes;
        off = (off + 255) & ~(size_t)255;
        return p;
    };

    _Float16* gw16    = (_Float16*)alloc((size_t)kFC * 384 * 2);
    _Float16* qw16    = (_Float16*)alloc((size_t)kD * kFC * 2);
    _Float16* kw16    = (_Float16*)alloc((size_t)kFC * kFC * 2);
    _Float16* vw16    = (_Float16*)alloc((size_t)kFC * kFC * 2);
    _Float16* ch1w16  = (_Float16*)alloc((size_t)2 * kFC * kFC * 2);
    _Float16* ch2w16  = (_Float16*)alloc((size_t)kFC * 2 * kFC * 2);
    _Float16* clsmw16 = (_Float16*)alloc((size_t)kFC * kFC * 2);
    _Float16* clsw16  = (_Float16*)alloc((size_t)2 * kFC * 2);
    _Float16* regmw16 = (_Float16*)alloc((size_t)kFC * kFC * 2);
    _Float16* regw16  = (_Float16*)alloc((size_t)74 * kFC * 2);
    _Float16* A_feat  = (_Float16*)alloc((size_t)kTOK * kKIN * 2);
    float*    feat32  = (float*)   alloc((size_t)kTOK * kFC * 4);
    _Float16* feat16  = (_Float16*)alloc((size_t)kTOK * kFC * 2);
    _Float16* q16     = (_Float16*)alloc((size_t)kTOK * kD * 2);
    _Float16* k16     = (_Float16*)alloc((size_t)kTOK * kFC * 2);
    _Float16* vT16    = (_Float16*)alloc((size_t)kB * kFC * kN * 2);
    _Float16* attn16  = (_Float16*)alloc((size_t)kB * kG * kN * kN * 2);
    _Float16* ctx16   = (_Float16*)alloc((size_t)kTOK * kFC * 2);
    _Float16* h1_16   = (_Float16*)alloc((size_t)kTOK * 2 * kFC * 2);
    _Float16* feat2_16= (_Float16*)alloc((size_t)kTOK * kFC * 2);
    _Float16* clsh16  = (_Float16*)alloc((size_t)kTOK * kFC * 2);
    _Float16* regh16  = (_Float16*)alloc((size_t)kTOK * kFC * 2);
    float*    cls32   = (float*)   alloc((size_t)kTOK * 2 * 4);
    float*    reg32   = (float*)   alloc((size_t)kTOK * 74 * 4);

    auto cvt = [&](const float* src, _Float16* dst, int n) {
        cvt_f16_kernel<<<(n + 255) / 256, 256, 0, stream>>>(src, dst, n);
    };
    cvt(gather_w, gw16,    kFC * 384);
    cvt(q_w,      qw16,    kD * kFC);
    cvt(k_w,      kw16,    kFC * kFC);
    cvt(v_w,      vw16,    kFC * kFC);
    cvt(ch1_w,    ch1w16,  2 * kFC * kFC);
    cvt(ch2_w,    ch2w16,  kFC * 2 * kFC);
    cvt(cls_m_w,  clsmw16, kFC * kFC);
    cvt(cls_w,    clsw16,  2 * kFC);
    cvt(reg_m_w,  regmw16, kFC * kFC);
    cvt(reg_w,    regw16,  74 * kFC);

    // 1) grid sample -> f16 A matrix [8192][2304]
    sample_kernel<<<kTOK, 256, 0, stream>>>(feat0, feat1, feat2, priors, z_emb, A_feat);

    // 2) grouped gather GEMM -> feat (fp32 + f16) [8192][192]
    gather_gemm<<<dim3(kTOK / 32, kFC / 32), 32, 0, stream>>>(A_feat, gw16, gather_b, feat32, feat16);

    // 3) q/k/v projections
    lin_gemm<<<dim3(kTOK / 32, 1), 32, 0, stream>>>(feat16, kFC, qw16, kFC, nullptr,
        kD, kFC, 0, 0, nullptr, 0, q16, kD, nullptr);
    lin_gemm<<<dim3(kTOK / 32, kFC / 32), 32, 0, stream>>>(feat16, kFC, kw16, kFC, nullptr,
        kFC, kFC, 0, 0, nullptr, 0, k16, kFC, nullptr);
    lin_gemm<<<dim3(kTOK / 32, kFC / 32), 32, 0, stream>>>(feat16, kFC, vw16, kFC, nullptr,
        kFC, kFC, 0, 1, nullptr, 0, vT16, 0, nullptr);   // vt=1: transposed store

    // 4) attention
    scores_gemm<<<dim3(kN / 32, kN / 32, kB * kG), 32, 0, stream>>>(q16, k16, attn_out);
    softmax_kernel<<<kB * kG * kN, 256, 0, stream>>>(attn_out, attn16);
    ctx_gemm<<<dim3(kN / 32, 1, kB * kG), 32, 0, stream>>>(attn16, vT16, ctx16);

    // 5) FFN with residual; fp32 result is fc_features (straight into d_out)
    lin_gemm<<<dim3(kTOK / 32, 2 * kFC / 32), 32, 0, stream>>>(ctx16, kFC, ch1w16, kFC, ch1_b,
        2 * kFC, kFC, 1, 0, nullptr, 0, h1_16, 2 * kFC, nullptr);
    lin_gemm<<<dim3(kTOK / 32, kFC / 32), 32, 0, stream>>>(h1_16, 2 * kFC, ch2w16, 2 * kFC, ch2_b,
        kFC, 2 * kFC, 0, 0, fc_out, kFC, feat2_16, kFC, feat32);

    // 6) heads
    lin_gemm<<<dim3(kTOK / 32, kFC / 32), 32, 0, stream>>>(feat2_16, kFC, clsmw16, kFC, cls_m_b,
        kFC, kFC, 1, 0, nullptr, 0, clsh16, kFC, nullptr);
    lin_gemm<<<dim3(kTOK / 32, 1), 32, 0, stream>>>(clsh16, kFC, clsw16, kFC, cls_b,
        2, kFC, 0, 0, cls32, 2, nullptr, 0, nullptr);
    lin_gemm<<<dim3(kTOK / 32, kFC / 32), 32, 0, stream>>>(feat2_16, kFC, regmw16, kFC, reg_m_b,
        kFC, kFC, 1, 0, nullptr, 0, regh16, kFC, nullptr);
    lin_gemm<<<dim3(kTOK / 32, 3), 32, 0, stream>>>(regh16, kFC, regw16, kFC, reg_b,
        74, kFC, 0, 0, reg32, 74, nullptr, 0, nullptr);

    // 7) predictions = [cls_logits, priors[:,:,2:] + reg]
    assemble_kernel<<<(kTOK * kPRI + 255) / 256, 256, 0, stream>>>(cls32, reg32, priors, pred_out);
}
